// MultiheadSelfAttention_37056977830576
// MI455X (gfx1250) — matrix operand order repaired and checked by reading, not compiled, not run on hardware
//
#include <hip/hip_runtime.h>
#include <hip/hip_bf16.h>

#define B_  2
#define S_  2048
#define D_  2048
#define H_  16
#define DH_ 128

#define USE_ASYNC 1

typedef __attribute__((ext_vector_type(16))) __bf16 v16bf;
typedef __attribute__((ext_vector_type(8)))  float  v8f;

#define WMMA_BF16(a, b, c) \
    __builtin_amdgcn_wmma_f32_16x16x32_bf16(false, (a), false, (b), (short)0, (c), false, false)

#define NEG_INF (-__builtin_inff())

// ---------- helpers ----------

__device__ __forceinline__ unsigned short f2bf(float f) {
    union { float f; unsigned u; } v; v.f = f;
    unsigned u = v.u;
    unsigned r = u + 0x7fffu + ((u >> 16) & 1u);   // round-to-nearest-even
    return (unsigned short)(r >> 16);
}
__device__ __forceinline__ unsigned pack2bf(float a, float b) {
    return (unsigned)f2bf(a) | ((unsigned)f2bf(b) << 16);
}

#if USE_ASYNC
// Async global->LDS copy, 16 bytes per lane. VDST = LDS byte address (flat shared
// pointer bits [31:0] are the LDS offset per the aperture mapping). ASYNCcnt-tracked.
__device__ __forceinline__ void async_copy_b128(void* lds_dst, const void* gsrc) {
    unsigned       laddr = (unsigned)(size_t)lds_dst;
    unsigned long long ga = (unsigned long long)(size_t)gsrc;
    asm volatile("global_load_async_to_lds_b128 %0, %1, off"
                 :: "v"(laddr), "v"(ga) : "memory");
}
__device__ __forceinline__ void wait_async0() {
    asm volatile("s_wait_asynccnt 0x0" ::: "memory");
}
#endif

// A-fragment (16x32 bf16, MxK): lane m = lane&15, hs = lane>>4.
// Element e<8 : K = e + hs*8 ; e>=8 : K = 16 + (e-8) + hs*8.  Pairs contiguous -> dword loads.
__device__ __forceinline__ v16bf lds_a_frag(const unsigned* A, int row, int kbase, int ldw) {
    union { v16bf v; unsigned u[8]; } r;
    const unsigned* p = A + row * ldw + ((kbase + ((threadIdx.x >> 4) & 1) * 8) >> 1);
#pragma unroll
    for (int i = 0; i < 4; i++) { r.u[i] = p[i]; r.u[4 + i] = p[8 + i]; }
    return r.v;
}

// B-fragment from Bt stored [n][k] (k contiguous): lane n = lane&15, same K order as A-fragment.
__device__ __forceinline__ v16bf lds_b_frag(const unsigned* Bt, int col, int kbase, int ldw) {
    union { v16bf v; unsigned u[8]; } r;
    const unsigned* p = Bt + col * ldw + ((kbase + ((threadIdx.x >> 4) & 1) * 8) >> 1);
#pragma unroll
    for (int i = 0; i < 4; i++) { r.u[i] = p[i]; r.u[4 + i] = p[8 + i]; }
    return r.v;
}

// ---------- kernel 1: fused QKV projection + RoPE ----------
// C = x @ W^T ; grid = (D/128, B*S/128, 3) ; block = 256 (8 waves, 4x2 wave grid)
// Register-pipelined: globals for tile k+1 load while tile k computes.

__global__ __launch_bounds__(256) void qkv_proj_kernel(
    const float* __restrict__ x,
    const float* __restrict__ Wq, const float* __restrict__ Wk, const float* __restrict__ Wv,
    const int* __restrict__ pos,
    unsigned short* __restrict__ Qb, unsigned short* __restrict__ Kb, unsigned short* __restrict__ Vb)
{
    __shared__ unsigned AsU[128 * 33];   // 128 x 64 bf16, padded (66 bf16 = 33 dwords/row)
    __shared__ unsigned BsU[128 * 33];

    const int z = blockIdx.z;
    const float* W = (z == 0) ? Wq : ((z == 1) ? Wk : Wv);
    unsigned short* out = (z == 0) ? Qb : ((z == 1) ? Kb : Vb);
    const bool doRope = (z < 2);

    const int mBase = blockIdx.y * 128;
    const int nBase = blockIdx.x * 128;
    const int t = threadIdx.x;
    const int lane = t & 31, l15 = lane & 15, hs = lane >> 4;
    const int w = t >> 5, wr = w >> 1, wc = w & 1;

    // per-thread staging slice: 8 float4 per matrix
    const int s_row = t >> 1;                 // 2 threads per row, 128 rows
    const int s_c4  = (t & 1) * 8;            // float4 column base (16 float4/row)

    v8f acc[2][4] = {};
    float4 ra[8], rb[8];

    // prologue: load tile 0 into registers
#pragma unroll
    for (int i = 0; i < 8; i++) {
        ra[i] = *(const float4*)(x + (size_t)(mBase + s_row) * D_ + (s_c4 + i) * 4);
        rb[i] = *(const float4*)(W + (size_t)(nBase + s_row) * D_ + (s_c4 + i) * 4);
    }

    for (int kb = 0; kb < D_; kb += 64) {
        // convert + store current tile to LDS
#pragma unroll
        for (int i = 0; i < 8; i++) {
            AsU[s_row * 33 + (s_c4 + i) * 2]     = pack2bf(ra[i].x, ra[i].y);
            AsU[s_row * 33 + (s_c4 + i) * 2 + 1] = pack2bf(ra[i].z, ra[i].w);
            BsU[s_row * 33 + (s_c4 + i) * 2]     = pack2bf(rb[i].x, rb[i].y);
            BsU[s_row * 33 + (s_c4 + i) * 2 + 1] = pack2bf(rb[i].z, rb[i].w);
        }
        __syncthreads();

        // issue next tile's global loads (overlap with WMMA below)
        if (kb + 64 < D_) {
#pragma unroll
            for (int i = 0; i < 8; i++) {
                ra[i] = *(const float4*)(x + (size_t)(mBase + s_row) * D_ + kb + 64 + (s_c4 + i) * 4);
                rb[i] = *(const float4*)(W + (size_t)(nBase + s_row) * D_ + kb + 64 + (s_c4 + i) * 4);
            }
        }

#pragma unroll
        for (int ks = 0; ks < 64; ks += 32) {
            v16bf a0 = lds_a_frag(AsU, wr * 32 + l15, ks, 33);
            v16bf a1 = lds_a_frag(AsU, wr * 32 + 16 + l15, ks, 33);
            v16bf bbf[4];
#pragma unroll
            for (int j = 0; j < 4; j++) bbf[j] = lds_b_frag(BsU, wc * 64 + j * 16 + l15, ks, 33);
#pragma unroll
            for (int j = 0; j < 4; j++) {
                acc[0][j] = WMMA_BF16(a0, bbf[j], acc[0][j]);
                acc[1][j] = WMMA_BF16(a1, bbf[j], acc[1][j]);
            }
        }
        __syncthreads();
    }

    // epilogue: optional interleaved RoPE, write bf16 in [B,H,S,DH]
#pragma unroll
    for (int i = 0; i < 2; i++) {
#pragma unroll
        for (int j = 0; j < 4; j++) {
            float* a = (float*)&acc[i][j];
            int ncol = nBase + wc * 64 + j * 16 + l15;
            int h = ncol >> 7, dh = ncol & (DH_ - 1);
#pragma unroll
            for (int r = 0; r < 8; r++) {
                int g  = mBase + wr * 32 + i * 16 + r + hs * 8;
                int bb = g >> 11, s = g & (S_ - 1);
                float val = a[r];
                float other = __shfl_xor(val, 1, 32);   // RoPE partner column (n ^ 1)
                if (doRope) {
                    int fi = dh >> 1;
                    float freq = __expf(-0.14391156f * (float)fi);  // (1/10000)^(2*fi/128)
                    float ang = (float)pos[s] * freq;
                    float sn, cs; __sincosf(ang, &sn, &cs);
                    val = (dh & 1) ? (other * sn + val * cs) : (val * cs - other * sn);
                }
                out[((size_t)(bb * H_ + h) * S_ + s) * DH_ + dh] = f2bf(val);
            }
        }
    }
}

// ---------- kernel 2: flash attention (causal, online softmax) ----------
// grid = (S/128, H, B) ; block = 256 ; wave w owns query rows [w*16, w*16+16)

__global__ __launch_bounds__(256) void attn_kernel(
    const unsigned short* __restrict__ Qb, const unsigned short* __restrict__ Kb,
    const unsigned short* __restrict__ Vb, unsigned short* __restrict__ Ob)
{
    __shared__ unsigned KsU[64 * 65];    // 64 keys x 128 dh bf16  (Bt for Q.K^T), pad->130 bf16
    __shared__ unsigned PsU[128 * 33];   // 128 q x 64 keys bf16   (A  for P.V)
    __shared__ unsigned VtU[128 * 33];   // 128 dh x 64 keys bf16  (Bt for P.V)

    const int qt = blockIdx.x, h = blockIdx.y, b = blockIdx.z;
    const int t = threadIdx.x;
    const int lane = t & 31, l15 = lane & 15, hs = lane >> 4;
    const int w = t >> 5;

    const size_t headOff = (size_t)(b * H_ + h) * S_ * DH_;

    // preload Q A-fragments for this wave's 16 rows (k = dh, 4 chunks of 32)
    v16bf qf[4];
    {
        const unsigned* qp = (const unsigned*)(Qb + headOff + (size_t)(qt * 128 + w * 16 + l15) * DH_);
#pragma unroll
        for (int ks = 0; ks < 4; ks++) {
            union { v16bf v; unsigned u[8]; } r;
            int k0 = (ks * 32 + hs * 8) >> 1;
#pragma unroll
            for (int i = 0; i < 4; i++) { r.u[i] = qp[k0 + i]; r.u[4 + i] = qp[k0 + 8 + i]; }
            qf[ks] = r.v;
        }
    }

    float m_old[8], l_sum[8];
    v8f o[8] = {};
#pragma unroll
    for (int r = 0; r < 8; r++) { m_old[r] = NEG_INF; l_sum[r] = 0.f; }

    const float rscale = 0.088388347648f;   // 1/sqrt(128)
    const int nkt = qt * 2 + 2;             // causal: keys up to qt*128+127

    for (int kt = 0; kt < nkt; kt++) {
        const int kBase = kt * 64;

        // stage K tile [key][dh] : pure bf16 copy -> async to LDS (ASYNCcnt)
        {
            const unsigned* src = (const unsigned*)(Kb + headOff + (size_t)kBase * DH_);
#pragma unroll
            for (int i = 0; i < 4; i++) {
                int idx = t + i * 256;          // 1024 x 16B
                int row = idx >> 4, c = idx & 15;
#if USE_ASYNC
                async_copy_b128(&KsU[row * 65 + c * 4], src + row * 64 + c * 4);
#else
                uint4 v = *(const uint4*)(src + row * 64 + c * 4);
                unsigned* d = &KsU[row * 65 + c * 4];
                d[0] = v.x; d[1] = v.y; d[2] = v.z; d[3] = v.w;
#endif
            }
        }
        // stage V tile transposed -> [dh][key] (paired dword repack)
        {
            const unsigned* src = (const unsigned*)(Vb + headOff + (size_t)kBase * DH_);
#pragma unroll
            for (int i = 0; i < 8; i++) {
                int idx = t + i * 256;          // 2048 key-pair x dword-col
                int kp = idx >> 6, c = idx & 63;
                unsigned d0 = src[(2 * kp) * 64 + c];
                unsigned d1 = src[(2 * kp + 1) * 64 + c];
                int dh = 2 * c;
                VtU[dh * 33 + kp]       = (d0 & 0xffffu) | (d1 << 16);
                VtU[(dh + 1) * 33 + kp] = (d0 >> 16) | (d1 & 0xffff0000u);
            }
        }
#if USE_ASYNC
        wait_async0();
#endif
        __syncthreads();

        // scores S = Q.K^T * rscale  (preload all B frags per k-step)
        v8f sc[4] = {};
#pragma unroll
        for (int ks = 0; ks < 4; ks++) {
            v16bf bbf[4];
#pragma unroll
            for (int j = 0; j < 4; j++) bbf[j] = lds_b_frag(KsU, j * 16 + l15, ks * 32, 65);
#pragma unroll
            for (int j = 0; j < 4; j++) sc[j] = WMMA_BF16(qf[ks], bbf[j], sc[j]);
        }

        // causal mask + online softmax (rows owned entirely by this wave)
#pragma unroll
        for (int r = 0; r < 8; r++) {
            const int qg = qt * 128 + w * 16 + r + hs * 8;
            float v = NEG_INF;
#pragma unroll
            for (int j = 0; j < 4; j++) {
                float* sp = (float*)&sc[j];
                int n = kBase + j * 16 + l15;
                float sv = sp[r] * rscale;
                if (n > qg) sv = NEG_INF;
                sp[r] = sv;
                v = fmaxf(v, sv);
            }
            v = fmaxf(v, __shfl_xor(v, 1, 16));
            v = fmaxf(v, __shfl_xor(v, 2, 16));
            v = fmaxf(v, __shfl_xor(v, 4, 16));
            v = fmaxf(v, __shfl_xor(v, 8, 16));
            float mn = fmaxf(m_old[r], v);
            float scale = __expf(m_old[r] - mn);
            float rs = 0.f;
#pragma unroll
            for (int j = 0; j < 4; j++) {
                float* sp = (float*)&sc[j];
                float p = __expf(sp[r] - mn);
                sp[r] = p;
                rs += p;
            }
            rs += __shfl_xor(rs, 1, 16);
            rs += __shfl_xor(rs, 2, 16);
            rs += __shfl_xor(rs, 4, 16);
            rs += __shfl_xor(rs, 8, 16);
            l_sum[r] = l_sum[r] * scale + rs;
#pragma unroll
            for (int j = 0; j < 8; j++) ((float*)&o[j])[r] *= scale;
            m_old[r] = mn;
        }

        // P -> LDS (bf16); rows owned by this wave (intra-wave LDS is in-order)
        {
            unsigned short* PsS = (unsigned short*)PsU;
#pragma unroll
            for (int j = 0; j < 4; j++) {
                float* sp = (float*)&sc[j];
#pragma unroll
                for (int r = 0; r < 8; r++) {
                    int m_loc = w * 16 + r + hs * 8;
                    PsS[m_loc * 66 + j * 16 + l15] = f2bf(sp[r]);
                }
            }
        }

        // O += P.V  (preload all 8 V B-frags per k-step)
#pragma unroll
        for (int ks = 0; ks < 64; ks += 32) {
            v16bf pa = lds_a_frag(PsU, w * 16 + l15, ks, 33);
            v16bf vbf[8];
#pragma unroll
            for (int j = 0; j < 8; j++) vbf[j] = lds_b_frag(VtU, j * 16 + l15, ks, 33);
#pragma unroll
            for (int j = 0; j < 8; j++) o[j] = WMMA_BF16(pa, vbf[j], o[j]);
        }
        __syncthreads();   // before next tile overwrites Ks/Vt
    }

    // finalize: O /= l ; write bf16 [B,S,D] (col = h*DH+dh)
    float inv[8];
#pragma unroll
    for (int r = 0; r < 8; r++) inv[r] = 1.0f / l_sum[r];
#pragma unroll
    for (int j = 0; j < 8; j++) {
        float* op = (float*)&o[j];
        int dh = j * 16 + l15;
#pragma unroll
        for (int r = 0; r < 8; r++) {
            int qg = qt * 128 + w * 16 + r + hs * 8;
            Ob[((size_t)(b * S_ + qg)) * D_ + h * DH_ + dh] = f2bf(op[r] * inv[r]);
        }
    }
}

// ---------- kernel 3: output projection  out = O @ Wo^T  (fp32 out) ----------
// A tile: async bf16 copy to LDS ; B tile: register-pipelined fp32->bf16.

__global__ __launch_bounds__(256) void out_proj_kernel(
    const unsigned short* __restrict__ Ob, const float* __restrict__ Wo,
    float* __restrict__ out)
{
    __shared__ unsigned AsU[128 * 33];
    __shared__ unsigned BsU[128 * 33];

    const int mBase = blockIdx.y * 128;
    const int nBase = blockIdx.x * 128;
    const int t = threadIdx.x;
    const int lane = t & 31, l15 = lane & 15, hs = lane >> 4;
    const int w = t >> 5, wr = w >> 1, wc = w & 1;

    const int s_row = t >> 1;
    const int s_c4  = (t & 1) * 8;

    v8f acc[2][4] = {};
    float4 rb[8];

#pragma unroll
    for (int i = 0; i < 8; i++)
        rb[i] = *(const float4*)(Wo + (size_t)(nBase + s_row) * D_ + (s_c4 + i) * 4);

    for (int kb = 0; kb < D_; kb += 64) {
        // A tile: 128x64 bf16 straight copy -> async to LDS
        {
            const unsigned* src = (const unsigned*)(Ob + (size_t)mBase * D_ + kb);
#pragma unroll
            for (int i = 0; i < 4; i++) {
                int idx = t + i * 256;          // 1024 x 16B
                int row = idx >> 3, c = idx & 7;
#if USE_ASYNC
                async_copy_b128(&AsU[row * 33 + c * 4], src + row * 1024 + c * 4);
#else
                uint4 v = *(const uint4*)(src + row * 1024 + c * 4);
                unsigned* d = &AsU[row * 33 + c * 4];
                d[0] = v.x; d[1] = v.y; d[2] = v.z; d[3] = v.w;
#endif
            }
        }
        // B tile: convert current regs, then prefetch next
#pragma unroll
        for (int i = 0; i < 8; i++) {
            BsU[s_row * 33 + (s_c4 + i) * 2]     = pack2bf(rb[i].x, rb[i].y);
            BsU[s_row * 33 + (s_c4 + i) * 2 + 1] = pack2bf(rb[i].z, rb[i].w);
        }
        if (kb + 64 < D_) {
#pragma unroll
            for (int i = 0; i < 8; i++)
                rb[i] = *(const float4*)(Wo + (size_t)(nBase + s_row) * D_ + kb + 64 + (s_c4 + i) * 4);
        }
#if USE_ASYNC
        wait_async0();
#endif
        __syncthreads();

#pragma unroll
        for (int ks = 0; ks < 64; ks += 32) {
            v16bf a0 = lds_a_frag(AsU, wr * 32 + l15, ks, 33);
            v16bf a1 = lds_a_frag(AsU, wr * 32 + 16 + l15, ks, 33);
            v16bf bbf[4];
#pragma unroll
            for (int j = 0; j < 4; j++) bbf[j] = lds_b_frag(BsU, wc * 64 + j * 16 + l15, ks, 33);
#pragma unroll
            for (int j = 0; j < 4; j++) {
                acc[0][j] = WMMA_BF16(a0, bbf[j], acc[0][j]);
                acc[1][j] = WMMA_BF16(a1, bbf[j], acc[1][j]);
            }
        }
        __syncthreads();
    }

#pragma unroll
    for (int i = 0; i < 2; i++) {
#pragma unroll
        for (int j = 0; j < 4; j++) {
            float* a = (float*)&acc[i][j];
            int ncol = nBase + wc * 64 + j * 16 + l15;
#pragma unroll
            for (int r = 0; r < 8; r++) {
                int g = mBase + wr * 32 + i * 16 + r + hs * 8;
                out[(size_t)g * D_ + ncol] = a[r];
            }
        }
    }
}

// ---------- launcher ----------

extern "C" void kernel_launch(void* const* d_in, const int* in_sizes, int n_in,
                              void* d_out, int out_size, void* d_ws, size_t ws_size,
                              hipStream_t stream)
{
    const float* x  = (const float*)d_in[0];
    const int* pos  = (const int*)d_in[1];
    const float* Wq = (const float*)d_in[2];
    const float* Wk = (const float*)d_in[3];
    const float* Wv = (const float*)d_in[4];
    const float* Wo = (const float*)d_in[5];

    const size_t NTOK = (size_t)B_ * S_ * D_;   // 8M bf16 elements per tensor
    unsigned short* Qb = (unsigned short*)d_ws;
    unsigned short* Kb = Qb + NTOK;
    unsigned short* Vb = Kb + NTOK;
    unsigned short* Ob = Vb + NTOK;

    dim3 blk(256);
    qkv_proj_kernel<<<dim3(D_ / 128, (B_ * S_) / 128, 3), blk, 0, stream>>>(
        x, Wq, Wk, Wv, pos, Qb, Kb, Vb);
    attn_kernel<<<dim3(S_ / 128, H_, B_), blk, 0, stream>>>(Qb, Kb, Vb, Ob);
    out_proj_kernel<<<dim3(D_ / 128, (B_ * S_) / 128), blk, 0, stream>>>(Ob, Wo, (float*)d_out);
}
